// GPT2_68375879352389
// MI455X (gfx1250) — matrix operand order repaired and checked by reading, not compiled, not run on hardware
//
#include <hip/hip_runtime.h>
#include <hip/hip_bf16.h>

#define D_MODEL 768
#define NHEAD   12
#define NLAYER  12
#define SEQ     1024
#define BATCH   2
#define HD      64
#define VOCAB   50257

typedef __attribute__((ext_vector_type(16))) __bf16 bf16x16;
typedef __attribute__((ext_vector_type(8)))  __bf16 bf16x8;
typedef __attribute__((ext_vector_type(8)))  float  f32x8;

// A-matrix (16x32 bf16) element->K map: lane holds row M=lane%16, hl=lane/16
__device__ __forceinline__ int amap(int e, int hl) {
  return ((e >> 3) << 4) + (hl << 3) + (e & 7);
}
// B-matrix (32x16 bf16) element->K map: lane holds col N=lane%16
__device__ __forceinline__ int bmap(int e, int hl) {
  return ((e >> 3) << 4) + (hl << 3) + (e & 7);
}

__device__ __forceinline__ f32x8 fzero8() {
  f32x8 z;
#pragma unroll
  for (int i = 0; i < 8; ++i) z[i] = 0.f;
  return z;
}

// ---------------------------------------------------------------- embedding
__global__ __launch_bounds__(256) void embed_kernel(
    const int* __restrict__ idx, const float* __restrict__ wte,
    const float* __restrict__ pos, float* __restrict__ x) {
  int row = blockIdx.x;          // b*T + t
  int t   = row % SEQ;
  int tok = idx[row];
  for (int c = threadIdx.x; c < D_MODEL; c += 256)
    x[(size_t)row * D_MODEL + c] =
        wte[(size_t)tok * D_MODEL + c] + pos[(size_t)t * D_MODEL + c];
}

// ---------------------------------------------------------------- layernorm
__global__ __launch_bounds__(256) void ln_kernel(
    const float* __restrict__ x, const float* __restrict__ g,
    const float* __restrict__ b, float* __restrict__ out) {
  __shared__ float red[256];
  int row = blockIdx.x;
  const float* xr = x + (size_t)row * D_MODEL;

  float s = 0.f;
  for (int c = threadIdx.x; c < D_MODEL; c += 256) s += xr[c];
  red[threadIdx.x] = s;
  __syncthreads();
  for (int off = 128; off > 0; off >>= 1) {
    if (threadIdx.x < off) red[threadIdx.x] += red[threadIdx.x + off];
    __syncthreads();
  }
  float mean = red[0] * (1.0f / D_MODEL);
  __syncthreads();

  float v = 0.f;
  for (int c = threadIdx.x; c < D_MODEL; c += 256) {
    float d = xr[c] - mean;
    v += d * d;
  }
  red[threadIdx.x] = v;
  __syncthreads();
  for (int off = 128; off > 0; off >>= 1) {
    if (threadIdx.x < off) red[threadIdx.x] += red[threadIdx.x + off];
    __syncthreads();
  }
  float rstd = rsqrtf(red[0] * (1.0f / D_MODEL) + 1e-5f);

  for (int c = threadIdx.x; c < D_MODEL; c += 256)
    out[(size_t)row * D_MODEL + c] = (xr[c] - mean) * rstd * g[c] + b[c];
}

// ------------------------------------------------- generic GEMM (bf16 WMMA)
// C[M,N] = act( A[M,K] @ W + bias ) + resid ;  W is [K,N] or (wtrans) [N,K]
// block = 256 threads = 8 waves; block tile 32(M) x 256(N); wave tile 16x64;
// K-step 64. LDS tiles stored in WMMA *fragment* layout so each lane pulls
// its 32-byte fragment with 2x ds_load_b128 (no per-element packing).
__global__ __launch_bounds__(256) void gemm_kernel(
    const float* __restrict__ A, const float* __restrict__ W,
    const float* __restrict__ bias, const float* __restrict__ resid,
    float* __restrict__ C, int M, int N, int K, int wtrans, int act_gelu) {
  // Af[waveM][khalf][lane] ; Bf[waveN][chunk][khalf][lane]
  __shared__ bf16x16 Af[2][2][32];          // 4 KB
  __shared__ bf16x16 Bf[4][4][2][32];       // 32 KB

  int mBase = blockIdx.y * 32;
  int nBase = blockIdx.x * 256;
  int tid   = threadIdx.x;
  int lane  = tid & 31, wid = tid >> 5;
  int waveM = wid >> 2, waveN = wid & 3;
  int hl    = lane >> 4, ml = lane & 15;

  __bf16* AfB = (__bf16*)Af;   // ((wm*2+kh)*32 + lane)*16 + e
  __bf16* BfB = (__bf16*)Bf;   // (((wn*4+c)*2+kh)*32 + lane)*16 + e

  f32x8 acc[4];
#pragma unroll
  for (int c = 0; c < 4; ++c) acc[c] = fzero8();

  for (int k0 = 0; k0 < K; k0 += 64) {
    // hint the next K-block of W toward the caches
    if (k0 + 64 < K) {
      const float* nw = wtrans ? &W[(size_t)nBase * K + k0 + 64]
                               : &W[(size_t)(k0 + 64) * N + nBase];
      __builtin_prefetch(nw + tid, 0, 0);
    }

    // stage A tile 32(M) x 64(K), fp32 -> bf16, into fragment layout
    for (int i = tid; i < 32 * 64; i += 256) {
      int r = i >> 6, kk = i & 63;                     // tid walks kk: coalesced
      float v = A[(size_t)(mBase + r) * K + k0 + kk];
      int wm = r >> 4, mr = r & 15;
      int kh = kk >> 5, k32 = kk & 31;
      int ah = (k32 >> 3) & 1;
      int ae = ((k32 >> 4) << 3) | (k32 & 7);
      AfB[((wm * 2 + kh) * 32 + ah * 16 + mr) * 16 + ae] = (__bf16)v;
    }
    // stage W tile 64(K) x 256(N) into fragment layout
    for (int it = 0; it < 64; ++it) {
      int i = it * 256 + tid;
      int kk, n;
      if (wtrans) { kk = i & 63; n = i >> 6; }         // tid walks kk: coalesced
      else        { n = i & 255; kk = i >> 8; }        // tid walks n : coalesced
      int gn = nBase + n;
      float v = 0.f;
      if (gn < N)
        v = wtrans ? W[(size_t)gn * K + k0 + kk]
                   : W[(size_t)(k0 + kk) * N + gn];
      int wn = n >> 6, c = (n >> 4) & 3, mr = n & 15;
      int kh = kk >> 5, k32 = kk & 31;
      int bh = (k32 >> 3) & 1, be = ((k32 >> 4) << 3) | (k32 & 7);
      BfB[(((wn * 4 + c) * 2 + kh) * 32 + bh * 16 + mr) * 16 + be] = (__bf16)v;
    }
    __syncthreads();

    bf16x16 a0 = Af[waveM][0][lane];                   // 2x ds_load_b128
    bf16x16 a1 = Af[waveM][1][lane];
#pragma unroll
    for (int c = 0; c < 4; ++c) {
      bf16x16 b0 = Bf[waveN][c][0][lane];
      bf16x16 b1 = Bf[waveN][c][1][lane];
      acc[c] = __builtin_amdgcn_wmma_f32_16x16x32_bf16(
          false, a0, false, b0, (short)0, acc[c], false, false);
      acc[c] = __builtin_amdgcn_wmma_f32_16x16x32_bf16(
          false, a1, false, b1, (short)0, acc[c], false, false);
    }
    __syncthreads();
  }

  // epilogue: bias / GELU(exact) / residual
#pragma unroll
  for (int c = 0; c < 4; ++c) {
#pragma unroll
    for (int r = 0; r < 8; ++r) {
      int m = mBase + waveM * 16 + hl * 8 + r;
      int n = nBase + waveN * 64 + c * 16 + ml;
      if (n < N) {
        float v = acc[c][r];
        if (bias) v += bias[n];
        if (act_gelu) v = 0.5f * v * (1.f + erff(v * 0.70710678118654752f));
        if (resid) v += resid[(size_t)m * N + n];
        C[(size_t)m * N + n] = v;
      }
    }
  }
}

// ---------------------------------------- flash attention (one wave / block)
// One wave handles 16 queries x all keys for one (batch, head), streaming
// softmax; 32 keys per step = 4 score WMMAs + 4 PV WMMAs.
__global__ __launch_bounds__(32) void attn_kernel(
    const float* __restrict__ qkv, float* __restrict__ o) {
  // P tile bounce buffer; row stride 40 elems (80 B) keeps the two 16-byte
  // fragment runs (hl*16, 32+hl*16) aligned for ds_load_b128.
  __shared__ __attribute__((aligned(32))) __bf16 Sp[16][40];

  int blk = blockIdx.x;
  int qi  = blk & 63;                       // SEQ/16 == 64
  int hh  = (blk >> 6) % NHEAD;
  int b   = blk / (64 * NHEAD);
  int lane = threadIdx.x & 31;
  int hl = lane >> 4, ml = lane & 15;
  int q0 = qi * 16;

  // Q fragments (16 x 64, pre-scaled by hd^-0.5 = 0.125)
  bf16x16 aq0, aq1;
  {
    size_t base = ((size_t)(b * SEQ + q0 + ml) * 3) * D_MODEL + hh * HD;
#pragma unroll
    for (int e = 0; e < 16; ++e) {
      int kd = amap(e, hl);
      aq0[e] = (__bf16)(qkv[base + kd] * 0.125f);
      aq1[e] = (__bf16)(qkv[base + kd + 32] * 0.125f);
    }
  }

  f32x8 accO[4];
#pragma unroll
  for (int c = 0; c < 4; ++c) accO[c] = fzero8();
  float mrun[8], lrun[8];
#pragma unroll
  for (int r = 0; r < 8; ++r) { mrun[r] = -3.0e38f; lrun[r] = 0.f; }

  int kend = q0 + 16;                       // exclusive causal key bound
  for (int j0 = 0; j0 < kend; j0 += 32) {
    // ---- scores S = Q @ K^T for keys [j0, j0+32)
    f32x8 s0 = fzero8(), s1 = fzero8();
    {
      size_t kb0 = ((size_t)(b * SEQ + j0 + ml) * 3 + 1) * D_MODEL + hh * HD;
      size_t kb1 = ((size_t)(b * SEQ + j0 + 16 + ml) * 3 + 1) * D_MODEL + hh * HD;
      bf16x16 b0, b1;
#pragma unroll
      for (int e = 0; e < 16; ++e) {
        int kd = bmap(e, hl);
        b0[e] = (__bf16)qkv[kb0 + kd];
        b1[e] = (__bf16)qkv[kb1 + kd];
      }
      s0 = __builtin_amdgcn_wmma_f32_16x16x32_bf16(false, aq0, false, b0,
                                                   (short)0, s0, false, false);
      s1 = __builtin_amdgcn_wmma_f32_16x16x32_bf16(false, aq0, false, b1,
                                                   (short)0, s1, false, false);
#pragma unroll
      for (int e = 0; e < 16; ++e) {
        int kd = bmap(e, hl) + 32;
        b0[e] = (__bf16)qkv[kb0 + kd];
        b1[e] = (__bf16)qkv[kb1 + kd];
      }
      s0 = __builtin_amdgcn_wmma_f32_16x16x32_bf16(false, aq1, false, b0,
                                                   (short)0, s0, false, false);
      s1 = __builtin_amdgcn_wmma_f32_16x16x32_bf16(false, aq1, false, b1,
                                                   (short)0, s1, false, false);
    }

    // ---- causal mask + online softmax (row = hl*8 + r, cols across lanes)
#pragma unroll
    for (int r = 0; r < 8; ++r) {
      int qg = q0 + hl * 8 + r;
      float v0 = ((j0 + ml) <= qg) ? s0[r] : -3.0e38f;
      float v1 = ((j0 + 16 + ml) <= qg) ? s1[r] : -3.0e38f;
      float tmax = fmaxf(v0, v1);
#pragma unroll
      for (int off = 8; off > 0; off >>= 1)
        tmax = fmaxf(tmax, __shfl_xor(tmax, off, 16));
      float mnew  = fmaxf(mrun[r], tmax);
      float scale = __expf(mrun[r] - mnew);
      float p0 = __expf(v0 - mnew);
      float p1 = __expf(v1 - mnew);
      float ps = p0 + p1;
#pragma unroll
      for (int off = 8; off > 0; off >>= 1) ps += __shfl_xor(ps, off, 16);
      lrun[r] = lrun[r] * scale + ps;
      mrun[r] = mnew;
#pragma unroll
      for (int c = 0; c < 4; ++c) accO[c][r] *= scale;
      Sp[hl * 8 + r][ml]      = (__bf16)p0;
      Sp[hl * 8 + r][16 + ml] = (__bf16)p1;
    }
    __syncthreads();

    // relayout P (C-frag) -> A-frag via LDS: two aligned 16B runs per lane
    bf16x16 pf;
    {
      const bf16x8* rowp = (const bf16x8*)&Sp[ml][0];
      bf16x8 lo = rowp[hl];          // elems  hl*8 .. hl*8+7   (K =  0..15 half)
      bf16x8 hi = rowp[2 + hl];      // elems 16+hl*8 ..        (K = 16..31 half)
#pragma unroll
      for (int e = 0; e < 8; ++e) { pf[e] = lo[e]; pf[8 + e] = hi[e]; }
    }
    __syncthreads();

    // ---- O += P @ V
    size_t vb = ((size_t)(b * SEQ + j0) * 3 + 2) * D_MODEL + hh * HD;
#pragma unroll
    for (int c = 0; c < 4; ++c) {
      bf16x16 vf;
#pragma unroll
      for (int e = 0; e < 16; ++e) {
        int key = bmap(e, hl);
        vf[e] = (__bf16)qkv[vb + (size_t)key * 3 * D_MODEL + c * 16 + ml];
      }
      accO[c] = __builtin_amdgcn_wmma_f32_16x16x32_bf16(
          false, pf, false, vf, (short)0, accO[c], false, false);
    }
  }

  // finalize: divide by row sums, write [B,T,D]
#pragma unroll
  for (int c = 0; c < 4; ++c) {
#pragma unroll
    for (int r = 0; r < 8; ++r) {
      int m = q0 + hl * 8 + r;
      o[(size_t)(b * SEQ + m) * D_MODEL + hh * HD + c * 16 + ml] =
          accO[c][r] / lrun[r];
    }
  }
}

// ---------------------------------------------------------------- launcher
extern "C" void kernel_launch(void* const* d_in, const int* in_sizes, int n_in,
                              void* d_out, int out_size, void* d_ws, size_t ws_size,
                              hipStream_t stream) {
  const int*   idx    = (const int*)  d_in[0];
  const float* wte    = (const float*)d_in[1];
  const float* pos    = (const float*)d_in[2];
  const float* ln1_g  = (const float*)d_in[3];
  const float* ln1_b  = (const float*)d_in[4];
  const float* qkv_w  = (const float*)d_in[5];
  const float* qkv_b  = (const float*)d_in[6];
  const float* proj_w = (const float*)d_in[7];
  const float* proj_b = (const float*)d_in[8];
  const float* ln2_g  = (const float*)d_in[9];
  const float* ln2_b  = (const float*)d_in[10];
  const float* fc1_w  = (const float*)d_in[11];
  const float* fc1_b  = (const float*)d_in[12];
  const float* fc2_w  = (const float*)d_in[13];
  const float* fc2_b  = (const float*)d_in[14];
  const float* lnf_g  = (const float*)d_in[15];
  const float* lnf_b  = (const float*)d_in[16];

  const int M = BATCH * SEQ;              // 2048 token rows
  float* ws   = (float*)d_ws;
  float* x    = ws;                                   // M*D
  float* h    = x    + (size_t)M * D_MODEL;           // M*D
  float* qkv  = h    + (size_t)M * D_MODEL;           // M*3D
  float* atto = qkv  + (size_t)M * 3 * D_MODEL;       // M*D
  float* mid  = atto + (size_t)M * D_MODEL;           // M*4D

  embed_kernel<<<M, 256, 0, stream>>>(idx, wte, pos, x);

  dim3 gQKV((3 * D_MODEL + 255) / 256, M / 32);
  dim3 gD  ((D_MODEL + 255) / 256,     M / 32);
  dim3 gFC1((4 * D_MODEL + 255) / 256, M / 32);
  dim3 gHead((VOCAB + 255) / 256,      M / 32);
  int  gAttn = BATCH * NHEAD * (SEQ / 16);

  for (int l = 0; l < NLAYER; ++l) {
    ln_kernel<<<M, 256, 0, stream>>>(x, ln1_g + (size_t)l * D_MODEL,
                                     ln1_b + (size_t)l * D_MODEL, h);
    gemm_kernel<<<gQKV, 256, 0, stream>>>(
        h, qkv_w + (size_t)l * D_MODEL * 3 * D_MODEL,
        qkv_b + (size_t)l * 3 * D_MODEL, nullptr, qkv,
        M, 3 * D_MODEL, D_MODEL, 0, 0);
    attn_kernel<<<gAttn, 32, 0, stream>>>(qkv, atto);
    gemm_kernel<<<gD, 256, 0, stream>>>(
        atto, proj_w + (size_t)l * D_MODEL * D_MODEL,
        proj_b + (size_t)l * D_MODEL, x, x,
        M, D_MODEL, D_MODEL, 0, 0);
    ln_kernel<<<M, 256, 0, stream>>>(x, ln2_g + (size_t)l * D_MODEL,
                                     ln2_b + (size_t)l * D_MODEL, h);
    gemm_kernel<<<gFC1, 256, 0, stream>>>(
        h, fc1_w + (size_t)l * D_MODEL * 4 * D_MODEL,
        fc1_b + (size_t)l * 4 * D_MODEL, nullptr, mid,
        M, 4 * D_MODEL, D_MODEL, 0, 1);
    gemm_kernel<<<gD, 256, 0, stream>>>(
        mid, fc2_w + (size_t)l * 4 * D_MODEL * D_MODEL,
        fc2_b + (size_t)l * D_MODEL, x, x,
        M, D_MODEL, 4 * D_MODEL, 0, 0);
  }

  ln_kernel<<<M, 256, 0, stream>>>(x, lnf_g, lnf_b, h);
  // tied head: logits = h @ wte^T  (wte is [V, D] -> transposed access)
  gemm_kernel<<<gHead, 256, 0, stream>>>(
      h, wte, nullptr, nullptr, (float*)d_out, M, VOCAB, D_MODEL, 1, 0);
}
